// AdaptiveDilatedConv2dLayer_34574486733152
// MI455X (gfx1250) — compile-verified, matrix operand order checked by reading, not run on hardware
//
#include <hip/hip_runtime.h>
#include <hip/hip_bf16.h>

typedef __attribute__((ext_vector_type(16))) __bf16 v16bf;
typedef __attribute__((ext_vector_type(2)))  __bf16 v2bf;
typedef __attribute__((ext_vector_type(8)))  float  v8f;

#define HO    126
#define WO    126
#define HIN   128
#define WIN   128
#define HW    (HIN * WIN)      // 16384
#define CIN   128
#define COUT  128
#define NKC   36               // K chunks: 1152 / 32
#define FRAG_ELEMS (NKC * 8 * 32 * 16)   // 147456 bf16 elements

// ---------------------------------------------------------------------------
// Kernel 1: bilinear upsample rates [32x32] -> rate_map [126x126]
// (align_corners=False semantics, matching the reference)
// ---------------------------------------------------------------------------
__global__ void adc_rate_kernel(const float* __restrict__ rates,
                                float* __restrict__ rate_map) {
    int idx = blockIdx.x * blockDim.x + threadIdx.x;
    if (idx >= HO * WO) return;
    int ho = idx / WO, wo = idx % WO;
    const float s = 32.0f / 126.0f;
    float sy = ((float)ho + 0.5f) * s - 0.5f;
    float sx = ((float)wo + 0.5f) * s - 0.5f;
    sy = fminf(fmaxf(sy, 0.0f), 31.0f);
    sx = fminf(fmaxf(sx, 0.0f), 31.0f);
    float y0f = floorf(sy), x0f = floorf(sx);
    int y0 = (int)y0f, x0 = (int)x0f;
    int y1 = (y0 + 1 < 31) ? y0 + 1 : 31;
    int x1 = (x0 + 1 < 31) ? x0 + 1 : 31;
    float wy = sy - y0f, wx = sx - x0f;
    float r00 = rates[y0 * 32 + x0], r01 = rates[y0 * 32 + x1];
    float r10 = rates[y1 * 32 + x0], r11 = rates[y1 * 32 + x1];
    rate_map[idx] = r00 * (1.0f - wy) * (1.0f - wx) + r01 * (1.0f - wy) * wx
                  + r10 * wy * (1.0f - wx)          + r11 * wy * wx;
}

// ---------------------------------------------------------------------------
// Kernel 2: pack weight [128][128][3][3] f32 into bf16 WMMA A-fragments.
// Fragment idx = ((kc*8 + m_tile)*32 + lane)*16 + i
// A layout (16-bit 16x32): lane L holds row M=L%16;
//   L<16 : K = {0..7, 16..23},  L>=16 : K = {8..15, 24..31}
//   element i -> k_local = (L<16?0:8) + (i<8 ? i : i+8)
// Global K index = tap*128 + c  (tap = kh*3+kw), chunk kc covers K kc*32..+31.
// ---------------------------------------------------------------------------
__global__ void adc_pack_kernel(const float* __restrict__ w,
                                __bf16* __restrict__ wfrag) {
    int idx = blockIdx.x * blockDim.x + threadIdx.x;
    if (idx >= FRAG_ELEMS) return;
    int i  = idx & 15;
    int L  = (idx >> 4) & 31;
    int mt = (idx >> 9) & 7;
    int kc = idx >> 12;
    int m  = L & 15;
    int kl = ((L >> 4) << 3) + (i & 7) + ((i >> 3) << 4);
    int tap = kc >> 2;                       // 0..8
    int c   = ((kc & 3) << 5) + kl;          // 0..127
    int cout = (mt << 4) + m;
    int kh = tap / 3, kw = tap % 3;
    wfrag[idx] = (__bf16)w[((cout * CIN + c) * 3 + kh) * 3 + kw];
}

// ---------------------------------------------------------------------------
// Kernel 3: main implicit-im2col GEMM with on-the-fly bilinear sampling.
// One block per (n, ho): M=128 (all Cout) x N=128 pixels (126 valid).
// 8 waves; wave = M-tile. B staged in LDS (double buffered).
// ---------------------------------------------------------------------------
__global__ void __launch_bounds__(256)
adc_main_kernel(const float* __restrict__ in,
                const __bf16* __restrict__ wfrag,
                const float* __restrict__ rate_map,
                const float* __restrict__ bias,
                float* __restrict__ out) {
    const int tid  = threadIdx.x;
    const int lane = tid & 31;
    const int wave = tid >> 5;
    const int n    = blockIdx.x / HO;
    const int ho   = blockIdx.x % HO;

    __shared__ __bf16 sB[2][8 * 32 * 16];   // 2 x 8KB B-fragment buffers

    // ---- producer-side per-thread state: one pixel slot, 16 channels ----
    const int   p    = tid & 127;           // pixel slot 0..127
    const int   cg   = tid >> 7;            // channel half 0/1
    const int   wo   = (p < WO) ? p : (WO - 1);
    const float live = (p < WO) ? 1.0f : 0.0f;   // dummy pixels contribute 0
    const float rate = rate_map[ho * WO + wo];
    const float* nin = in + (size_t)n * CIN * HW;
    const int   nt_p = p >> 4;              // producer's N-tile
    const int   ncol = p & 15;

    int   o00 = 0, o01 = 0, o10 = 0, o11 = 0;
    float W00 = 0.f, W01 = 0.f, W10 = 0.f, W11 = 0.f;
    int   cur_tap = -1;

    auto set_tap = [&](int tap) {
        int kh = tap / 3, kw = tap % 3;
        float fy = (float)ho + (float)kh * rate;   // >= 0 always
        float fx = (float)wo + (float)kw * rate;   // >= 0 always
        float y0f = floorf(fy), x0f = floorf(fx);
        int iy0 = (int)y0f, ix0 = (int)x0f;
        int iy1 = iy0 + 1,  ix1 = ix0 + 1;
        float wy = fy - y0f, wx = fx - x0f;
        float my0 = (iy0 <= HIN - 1) ? 1.0f : 0.0f;
        float my1 = (iy1 <= HIN - 1) ? 1.0f : 0.0f;
        float mx0 = (ix0 <= WIN - 1) ? 1.0f : 0.0f;
        float mx1 = (ix1 <= WIN - 1) ? 1.0f : 0.0f;
        int cy0 = (iy0 < HIN - 1) ? iy0 : HIN - 1;
        int cy1 = (iy1 < HIN - 1) ? iy1 : HIN - 1;
        int cx0 = (ix0 < WIN - 1) ? ix0 : WIN - 1;
        int cx1 = (ix1 < WIN - 1) ? ix1 : WIN - 1;
        o00 = cy0 * WIN + cx0;  o01 = cy0 * WIN + cx1;
        o10 = cy1 * WIN + cx0;  o11 = cy1 * WIN + cx1;
        W00 = (1.0f - wy) * (1.0f - wx) * my0 * mx0 * live;
        W01 = (1.0f - wy) * wx          * my0 * mx1 * live;
        W10 = wy * (1.0f - wx)          * my1 * mx0 * live;
        W11 = wy * wx                   * my1 * mx1 * live;
    };

    auto stage = [&](int kc, int buf) {
        int tap = kc >> 2;
        if (tap != cur_tap) { set_tap(tap); cur_tap = tap; }
        const int cbase = ((kc & 3) << 5) + (cg << 4);
        const int klhi  = (cg << 4);             // k_local bit4 per channel half
#pragma unroll
        for (int j = 0; j < 16; j += 2) {
            const float* ch0 = nin + (size_t)(cbase + j) * HW;
            const float* ch1 = ch0 + HW;
            float v0 = W00 * ch0[o00] + W01 * ch0[o01] + W10 * ch0[o10] + W11 * ch0[o11];
            float v1 = W00 * ch1[o00] + W01 * ch1[o01] + W10 * ch1[o10] + W11 * ch1[o11];
            int kl = klhi + j;                                  // even
            int bb = (kl >> 3) & 1;
            int ii = (kl & 7) + ((kl & 16) >> 1);               // even -> 4B aligned
            int Lw = ncol + (bb << 4);
            v2bf pr; pr.x = (__bf16)v0; pr.y = (__bf16)v1;
            *(v2bf*)&sB[buf][(nt_p * 32 + Lw) * 16 + ii] = pr;
        }
    };

    v8f acc[8] = {};

    stage(0, 0);
    __syncthreads();

    for (int kc = 0; kc < NKC; ++kc) {
        // Issue this chunk's operand loads FIRST so they overlap with the
        // sampling burst below. Distinct registers per B fragment so the 16
        // ds_load_b128 stay in flight together (staggered s_wait_dscnt).
        const v16bf a = *(const v16bf*)(wfrag + ((size_t)(kc * 8 + wave) * 32 + lane) * 16);
        const __bf16* bbase = &sB[kc & 1][0];
        v16bf b0 = *(const v16bf*)(bbase + (0 * 32 + lane) * 16);
        v16bf b1 = *(const v16bf*)(bbase + (1 * 32 + lane) * 16);
        v16bf b2 = *(const v16bf*)(bbase + (2 * 32 + lane) * 16);
        v16bf b3 = *(const v16bf*)(bbase + (3 * 32 + lane) * 16);
        v16bf b4 = *(const v16bf*)(bbase + (4 * 32 + lane) * 16);
        v16bf b5 = *(const v16bf*)(bbase + (5 * 32 + lane) * 16);
        v16bf b6 = *(const v16bf*)(bbase + (6 * 32 + lane) * 16);
        v16bf b7 = *(const v16bf*)(bbase + (7 * 32 + lane) * 16);

        if (kc + 1 < NKC) {
            // near-cache (WGP) prefetch of next A fragment
            __builtin_prefetch(wfrag + ((size_t)((kc + 1) * 8 + wave) * 32 + lane) * 16, 0, 3);
            stage(kc + 1, (kc + 1) & 1);
        }

        acc[0] = __builtin_amdgcn_wmma_f32_16x16x32_bf16(false, a, false, b0, (short)0, acc[0], false, false);
        acc[1] = __builtin_amdgcn_wmma_f32_16x16x32_bf16(false, a, false, b1, (short)0, acc[1], false, false);
        acc[2] = __builtin_amdgcn_wmma_f32_16x16x32_bf16(false, a, false, b2, (short)0, acc[2], false, false);
        acc[3] = __builtin_amdgcn_wmma_f32_16x16x32_bf16(false, a, false, b3, (short)0, acc[3], false, false);
        acc[4] = __builtin_amdgcn_wmma_f32_16x16x32_bf16(false, a, false, b4, (short)0, acc[4], false, false);
        acc[5] = __builtin_amdgcn_wmma_f32_16x16x32_bf16(false, a, false, b5, (short)0, acc[5], false, false);
        acc[6] = __builtin_amdgcn_wmma_f32_16x16x32_bf16(false, a, false, b6, (short)0, acc[6], false, false);
        acc[7] = __builtin_amdgcn_wmma_f32_16x16x32_bf16(false, a, false, b7, (short)0, acc[7], false, false);

        __syncthreads();
    }

    // ---- epilogue: C layout lane L, vgpr r -> (M = r + 8*(L/16), N = L%16) ----
    const int m_hi  = (lane >> 4) << 3;   // 0 or 8
    const int ncolC = lane & 15;
#pragma unroll
    for (int nt = 0; nt < 8; ++nt) {
        int wo2 = nt * 16 + ncolC;
        if (wo2 < WO) {
#pragma unroll
            for (int r = 0; r < 8; ++r) {
                int cout = (wave << 4) + m_hi + r;
                out[(((size_t)n * COUT + cout) * HO + ho) * WO + wo2] =
                    acc[nt][r] + bias[cout];
            }
        }
    }
}

// ---------------------------------------------------------------------------
extern "C" void kernel_launch(void* const* d_in, const int* in_sizes, int n_in,
                              void* d_out, int out_size, void* d_ws, size_t ws_size,
                              hipStream_t stream) {
    const float* inputs = (const float*)d_in[0];   // [8,128,128,128]
    const float* weight = (const float*)d_in[1];   // [128,128,3,3]
    const float* rates  = (const float*)d_in[2];   // [1,1,32,32]
    const float* bias   = (const float*)d_in[3];   // [128]
    float* out = (float*)d_out;                    // [8,128,126,126]

    float*  rate_map = (float*)d_ws;                              // 63.5 KB
    __bf16* wfrag    = (__bf16*)((char*)d_ws + 65536);            // 288 KB

    adc_rate_kernel<<<(HO * WO + 255) / 256, 256, 0, stream>>>(rates, rate_map);
    adc_pack_kernel<<<(FRAG_ELEMS + 255) / 256, 256, 0, stream>>>(weight, wfrag);
    adc_main_kernel<<<8 * HO, 256, 0, stream>>>(inputs, wfrag, rate_map, bias, out);
}
